// VectorQuantizer_38319698215672
// MI455X (gfx1250) — matrix operand order repaired and checked by reading, not compiled, not run on hardware
//
#include <hip/hip_runtime.h>
#include <hip/hip_bf16.h>

typedef __attribute__((ext_vector_type(16))) __bf16 v16bf;
typedef __attribute__((ext_vector_type(8)))  float  v8f;
typedef __attribute__((ext_vector_type(4)))  float  v4f;

#define D_DIM   128
#define K_CODES 8192
#define N_VEC   16384          // 8 * 2048
#define N_TILES (K_CODES / 16) // 512

// ---------------------------------------------------------------- kernel 1:
// per-code squared norms; coalesced (thread k walks column k). Also zero loss.
__global__ __launch_bounds__(256) void vq_enorm(const float* __restrict__ emb,
                                                float* __restrict__ enorm,
                                                float* __restrict__ lossAcc) {
    int k = blockIdx.x * 256 + threadIdx.x;      // 0..8191
    float s = 0.f;
    for (int d = 0; d < D_DIM; ++d) { float v = emb[d * K_CODES + k]; s += v * v; }
    enorm[k] = s;
    if (k == 0) *lossAcc = 0.f;
}

// ---------------------------------------------------------------- kernel 2:
// pack embeddings into WMMA B-fragment order (bf16). Fragment f=(nt*4+kb):
// lane l holds column nt*16+(l&15), K = kb*32 + ((l&16)?16:0) + i, i=0..15.
// Stored as 32 contiguous bytes per lane -> coalesced 1KB loads in hot loop.
__global__ __launch_bounds__(256) void vq_pack_b(const float* __restrict__ emb,
                                                 __bf16* __restrict__ bfrag) {
    int t    = blockIdx.x * 256 + threadIdx.x;   // 65536 threads
    int frag = t >> 5;                           // 0..2047
    int lane = t & 31;
    int nt = frag >> 2, kb = frag & 3;
    int col = nt * 16 + (lane & 15);
    int d0  = kb * 32 + ((lane & 16) ? 16 : 0);
    __bf16* dst = bfrag + ((size_t)(frag * 32 + lane)) * 16;
    #pragma unroll
    for (int i = 0; i < 16; ++i)
        dst[i] = (__bf16)emb[(size_t)(d0 + i) * K_CODES + col];
}

// ---------------------------------------------------------------- kernel 3:
// main: per 16-row tile, sweep all 8192 codes with bf16 WMMA, fused argmin of
// score = ||e||^2 - 2*x.e (||x||^2 constant per row -> dropped).
__global__ __launch_bounds__(256) void vq_argmin_wmma(const float*  __restrict__ inputs,
                                                      const float*  __restrict__ enorm,
                                                      const v16bf*  __restrict__ bfrag,
                                                      int*          __restrict__ outIdx) {
    __shared__ __bf16 aLds[16 * D_DIM];          // 4 KB bf16 A tile
    __shared__ float  redD[8][16];
    __shared__ int    redI[8][16];

    const int tid  = threadIdx.x;
    const int m0   = blockIdx.x * 16;
    const int wave = tid >> 5;
    const int lane = tid & 31;
    const int half = lane >> 4;                  // 0: lanes 0-15, 1: lanes 16-31
    const int l16  = lane & 15;

    // stage A tile (16x128 f32 -> bf16 in LDS)
    #pragma unroll
    for (int i = 0; i < 8; ++i) {
        int e = tid * 8 + i;                     // 0..2047
        aLds[e] = (__bf16)inputs[(size_t)m0 * D_DIM + e];
    }
    __syncthreads();

    // build 4 A fragments in contiguous k order (same as packed B):
    // lanes 0-15: K = {0..15}; lanes 16-31: K = {16..31} (+32*kb)
    v16bf afrag[4];
    const int kb8 = half * 16;
    #pragma unroll
    for (int kb = 0; kb < 4; ++kb) {
        #pragma unroll
        for (int i = 0; i < 8; ++i) {
            afrag[kb][i]     = aLds[l16 * D_DIM + kb * 32 + kb8 + i];
            afrag[kb][8 + i] = aLds[l16 * D_DIM + kb * 32 + kb8 + 8 + i];
        }
    }

    float bestD[8];
    int   bestI[8];
    #pragma unroll
    for (int j = 0; j < 8; ++j) { bestD[j] = 3.4e38f; bestI[j] = 0; }

    // each wave handles 64 of the 512 N-tiles
    for (int nt = wave; nt < N_TILES; nt += 8) {
        float en = enorm[nt * 16 + l16];         // this lane's column norm
        const v16bf* bp = bfrag + (size_t)(nt * 4) * 32 + lane;
        v8f c = {};
        #pragma unroll
        for (int kb = 0; kb < 4; ++kb) {
            v16bf b = bp[(size_t)kb * 32];
            c = __builtin_amdgcn_wmma_f32_16x16x32_bf16(
                    false, afrag[kb], false, b, (short)0, c, false, false);
        }
        // C layout: lane<16 -> VGPR j = (M=j, N=lane); lane>=16 -> (M=j+8, N=lane-16)
        int col = nt * 16 + l16;
        #pragma unroll
        for (int j = 0; j < 8; ++j) {
            float dist = en - 2.0f * c[j];
            if (dist < bestD[j]) { bestD[j] = dist; bestI[j] = col; }
        }
    }

    // reduce across the 16 lanes of each half (rows 0-7 / 8-15)
    #pragma unroll
    for (int j = 0; j < 8; ++j) {
        float bd = bestD[j]; int bi = bestI[j];
        #pragma unroll
        for (int off = 8; off >= 1; off >>= 1) {
            float od = __shfl_xor(bd, off, 16);
            int   oi = __shfl_xor(bi, off, 16);
            if (od < bd || (od == bd && oi < bi)) { bd = od; bi = oi; }
        }
        if (l16 == 0) { int row = j + half * 8; redD[wave][row] = bd; redI[wave][row] = bi; }
    }
    __syncthreads();

    // combine the 8 waves
    if (tid < 16) {
        float bd = redD[0][tid]; int bi = redI[0][tid];
        #pragma unroll
        for (int w = 1; w < 8; ++w) {
            float od = redD[w][tid]; int oi = redI[w][tid];
            if (od < bd || (od == bd && oi < bi)) { bd = od; bi = oi; }
        }
        outIdx[m0 + tid] = bi;
    }
}

// ---------------------------------------------------------------- kernel 4:
// quantized gather (straight-through output == gathered code), loss partials,
// indices-as-float. 2 rows per 256-thread block.
__global__ __launch_bounds__(256) void vq_quantize(const float* __restrict__ inputs,
                                                   const float* __restrict__ emb,
                                                   const int*   __restrict__ idx,
                                                   float*       __restrict__ qOut,
                                                   float*       __restrict__ idxOut,
                                                   float*       __restrict__ lossAcc) {
    __shared__ float red[256];
    int tid = threadIdx.x;
    int r   = blockIdx.x * 2 + (tid >> 7);
    int d   = tid & 127;
    int k   = idx[r];
    float q = emb[(size_t)d * K_CODES + k];      // column gather; L2-resident
    float x = inputs[(size_t)r * D_DIM + d];
    qOut[(size_t)r * D_DIM + d] = q;             // x + (q - x) == q
    if (d == 0) idxOut[r] = (float)k;
    float diff = q - x;
    red[tid] = diff * diff;
    __syncthreads();
    for (int s = 128; s > 0; s >>= 1) {
        if (tid < s) red[tid] += red[tid + s];
        __syncthreads();
    }
    if (tid == 0) atomicAdd(lossAcc, red[0]);
}

// ---------------------------------------------------------------- kernel 5:
// one-hot encodings: 512 MB of b128 stores (the bandwidth limiter).
// Each block owns a 4096-float slice of ONE row -> idx[row] is wave-uniform
// (scalar load). Each thread writes 4 x 16B with non-temporal hint so this
// write-once stream doesn't sweep the 192 MB L2 (keeps embeddings / fragments
// resident for the other kernels).
__global__ __launch_bounds__(256) void vq_onehot(const int* __restrict__ idx,
                                                 float*     __restrict__ enc) {
    int blk = blockIdx.x;
    int row = blk >> 1;                            // 2 blocks per row
    int k   = idx[row];                            // uniform -> s_load
    // this thread's 16-float chunk within the row
    int c0  = ((blk & 1) * 256 + threadIdx.x) * 16;      // column of first float
    v4f* p  = (v4f*)(enc + (size_t)row * K_CODES + c0);
    #pragma unroll
    for (int q = 0; q < 4; ++q) {
        int cb = c0 + q * 4;
        v4f v = {0.f, 0.f, 0.f, 0.f};
        if (k >= cb && k < cb + 4) v[k - cb] = 1.f;
        __builtin_nontemporal_store(v, p + q);
    }
}

// ---------------------------------------------------------------- kernel 6:
__global__ void vq_loss_final(const float* __restrict__ lossAcc,
                              float*       __restrict__ lossOut) {
    // q_latent_loss + 0.25 * e_latent_loss, both numerically mean((q-x)^2)
    *lossOut = 1.25f * (*lossAcc) / (float)(N_VEC * D_DIM);
}

extern "C" void kernel_launch(void* const* d_in, const int* in_sizes, int n_in,
                              void* d_out, int out_size, void* d_ws, size_t ws_size,
                              hipStream_t stream) {
    const float* inputs = (const float*)d_in[0];   // [8,2048,128] f32
    const float* emb    = (const float*)d_in[1];   // [128,8192]  f32

    float* out      = (float*)d_out;
    float* q_out    = out;                                           // 2,097,152
    float* enc_out  = out + 2097152;                                 // 134,217,728
    float* idx_out  = out + 2097152 + 134217728;                     // 16,384
    float* loss_out = out + 2097152 + 134217728 + 16384;             // 1

    char*   ws      = (char*)d_ws;
    float*  enorm   = (float*)ws;                  // 32 KB
    int*    idx     = (int*)(ws + 32768);          // 64 KB
    float*  lossAcc = (float*)(ws + 98304);        // 4 B
    __bf16* bfrag   = (__bf16*)(ws + 131072);      // 2 MB packed B fragments

    vq_enorm      <<<K_CODES / 256, 256, 0, stream>>>(emb, enorm, lossAcc);
    vq_pack_b     <<<(N_TILES * 4 * 32) / 256, 256, 0, stream>>>(emb, bfrag);
    vq_argmin_wmma<<<N_VEC / 16, 256, 0, stream>>>(inputs, enorm, (const v16bf*)bfrag, idx);
    vq_quantize   <<<N_VEC / 2, 256, 0, stream>>>(inputs, emb, idx, q_out, idx_out, lossAcc);
    vq_onehot     <<<N_VEC * 2, 256, 0, stream>>>(idx, enc_out);
    vq_loss_final <<<1, 1, 0, stream>>>(lossAcc, loss_out);
}
